// RelationalMessagePassingModule_65377992180482
// MI455X (gfx1250) — matrix operand order repaired and checked by reading, not compiled, not run on hardware
//
#include <hip/hip_runtime.h>
#include <hip/hip_bf16.h>

typedef _Float16 v16h __attribute__((ext_vector_type(16)));
typedef _Float16 v8h  __attribute__((ext_vector_type(8)));
typedef _Float16 v4h  __attribute__((ext_vector_type(4)));
typedef float    v8f  __attribute__((ext_vector_type(8)));

#define SMOOTH 12.0f

// Branch-free mish: with e = exp(x),
// tanh(softplus(x)) = ((1+e)^2 - 1)/((1+e)^2 + 1) = (e*e + 2e)/(e*e + 2e + 2)
// Clamp exp arg at 30: then t/(t+2) == 1 in f32 => mish(x) == x.
// Denominator >= 2, so a single v_rcp_f32 is safe (no zero/denorm cases).
__device__ __forceinline__ float mishf(float x) {
    float e = __expf(fminf(x, 30.0f));
    float t = __builtin_fmaf(e, e, 2.0f * e);
    return x * t * __builtin_amdgcn_rcpf(t + 2.0f);
}

// Order-preserving float<->uint encoding so unsigned atomicMax == float max.
// enc is strictly increasing in float order; enc value 0 is below every real
// float, so hipMemsetAsync(mx, 0, ...) initializes to "-inf".
__device__ __forceinline__ unsigned fenc(float f) {
    unsigned u = __float_as_uint(f);
    return (u & 0x80000000u) ? ~u : (u | 0x80000000u);
}
__device__ __forceinline__ float fdec(unsigned e) {
    unsigned u = (e & 0x80000000u) ? (e ^ 0x80000000u) : ~e;
    return __uint_as_float(u);
}

// --- A fragment: 16x32 f16 tile from row-major [16][K] f16 LDS buffer ---
// lanes 0-15: row=lane, elems 0..7 = K kb+0..7,  elems 8..15 = K kb+16..23
// lanes16-31: row=lane-16, elems 0..7 = K kb+8..15, elems 8..15 = K kb+24..31
__device__ __forceinline__ v16h load_a(const _Float16* buf, int K, int mrow,
                                       int half_id, int kb) {
    union { v16h v; v8h h2[2]; } au;
    au.h2[0] = *(const v8h*)(buf + mrow * K + kb + half_id * 8);
    au.h2[1] = *(const v8h*)(buf + mrow * K + kb + 16 + half_id * 8);
    return au.v;
}

// --- B fragment: 32x16 f16 tile from f16 weight stored [N][K] row-major ---
// lane holds column (nb + lane&15), elems 0..7 = K kb+half*8+0..7, elems 8..15 = K kb+16+half*8+0..7
__device__ __forceinline__ v16h load_b(const _Float16* wT, int K, int col,
                                       int half_id, int kb) {
    union { v16h v; v8h h2[2]; } bu;
    bu.h2[0] = *(const v8h*)(wT + (long)col * K + kb + half_id * 8);
    bu.h2[1] = *(const v8h*)(wT + (long)col * K + kb + 16 + half_id * 8);
    return bu.v;
}

__device__ __forceinline__ v8f wmma_f16(v16h a, v16h b, v8f c) {
    return __builtin_amdgcn_wmma_f32_16x16x32_f16(false, a, false, b,
                                                  (short)0, c, false, false);
}

// Transpose + convert fp32 [K][N] -> fp16 [N][K]
__global__ void wconv_kernel(const float* __restrict__ in,
                             _Float16* __restrict__ out, int K, int N) {
    int t = blockIdx.x * blockDim.x + threadIdx.x;
    if (t >= K * N) return;
    int k = t / N, n = t % N;
    out[(long)n * K + k] = (_Float16)in[t];
}

// Relation message kernel: 4 waves / block, each wave owns 16 edge rows.
// e = gather(h, idx) [16 x K], m = e + mish(e@wi+bi)@wo + bo,
// store m (f16) and atomicMax encoded segment maxima.
template <int ARITY>
__global__ void __launch_bounds__(128)
msg_kernel(const float* __restrict__ h,
           const int* __restrict__ idx,
           const _Float16* __restrict__ wiT,
           const float* __restrict__ bi,
           const _Float16* __restrict__ woT,
           const float* __restrict__ bo,
           _Float16* __restrict__ msg,
           unsigned* __restrict__ mx,
           int nedges) {
    constexpr int K  = ARITY * 64;
    constexpr int NT = K / 16;   // output column tiles
    constexpr int KS = K / 32;   // K accumulation steps
    __shared__ _Float16 lds_e[4 * 16 * K];
    __shared__ _Float16 lds_m[4 * 16 * K];
    __shared__ int      lds_idx[4 * 16 * ARITY];

    const int  wave = threadIdx.x >> 5;
    const int  lane = threadIdx.x & 31;
    const long base = (long)blockIdx.x * 64 + wave * 16;

    _Float16* my_e   = lds_e + wave * 16 * K;
    _Float16* my_m   = lds_m + wave * 16 * K;
    int*      my_idx = lds_idx + wave * 16 * ARITY;

    // Stage indices for this wave's 16 edge rows
    for (int t = lane; t < 16 * ARITY; t += 32) {
        long gi = base * ARITY + t;
        my_idx[t] = (gi < (long)nedges * ARITY) ? idx[gi] : 0;
    }
    __syncthreads();

    // Gather e rows from h (L2-resident), convert to f16 into LDS
    for (int t = lane; t < 16 * (K / 4); t += 32) {
        int r = t / (K / 4);
        int c = (t % (K / 4)) * 4;
        int hrow = my_idx[r * ARITY + (c >> 6)];
        const float4 v = *(const float4*)(h + (long)hrow * 64 + (c & 63));
        _Float16* d = &my_e[r * K + c];
        d[0] = (_Float16)v.x; d[1] = (_Float16)v.y;
        d[2] = (_Float16)v.z; d[3] = (_Float16)v.w;
    }
    __syncthreads();

    const int half_id = lane >> 4;
    const int nlane   = lane & 15;

    // Hoist the whole 16xK A panel into registers (reused for all N tiles)
    v16h afrag[KS];
#pragma unroll
    for (int ks = 0; ks < KS; ++ks)
        afrag[ks] = load_a(my_e, K, nlane, half_id, ks * 32);

    // GEMM1: hmid = mish(e @ wi + bi)
    for (int nt = 0; nt < NT; ++nt) {
        const int col = nt * 16 + nlane;
        const float b0 = bi[col];
        v8f acc;
#pragma unroll
        for (int j = 0; j < 8; ++j) acc[j] = b0;
#pragma unroll
        for (int ks = 0; ks < KS; ++ks)
            acc = wmma_f16(afrag[ks], load_b(wiT, K, col, half_id, ks * 32), acc);
#pragma unroll
        for (int j = 0; j < 8; ++j) {
            int row = j + 8 * half_id;
            my_m[row * K + col] = (_Float16)mishf(acc[j]);
        }
    }
    __syncthreads();

    // Hoist hmid A panel
    v16h mfrag[KS];
#pragma unroll
    for (int ks = 0; ks < KS; ++ks)
        mfrag[ks] = load_a(my_m, K, nlane, half_id, ks * 32);

    // GEMM2: m = e + hmid @ wo + bo ; emit messages + segment atomic max
    for (int nt = 0; nt < NT; ++nt) {
        const int col = nt * 16 + nlane;
        const float b0 = bo[col];
        v8f acc;
#pragma unroll
        for (int j = 0; j < 8; ++j) acc[j] = b0;
#pragma unroll
        for (int ks = 0; ks < KS; ++ks)
            acc = wmma_f16(mfrag[ks], load_b(woT, K, col, half_id, ks * 32), acc);

        const int sub = col >> 6;      // message slot within the edge row
        const int cm  = col & 63;      // column within E=64
#pragma unroll
        for (int j = 0; j < 8; ++j) {
            int  row  = j + 8 * half_id;
            long erow = base + row;
            if (erow < nedges) {
                float val = acc[j] + (float)my_e[row * K + col];
                long  seg = my_idx[row * ARITY + sub];
                msg[(erow * ARITY + sub) * 64 + cm] = (_Float16)val;
                atomicMax(&mx[seg * 64 + cm], fenc(val));
            }
        }
    }
}

// Second segment pass: s[seg] += exp(SMOOTH * (m - mx[seg])), 4 cols/thread
__global__ void expsum_kernel(const _Float16* __restrict__ msg,
                              const int* __restrict__ idx,
                              const unsigned* __restrict__ mx,
                              float* __restrict__ s, int nmsg) {
    int g = blockIdx.x * blockDim.x + threadIdx.x;
    if (g >= nmsg * 16) return;
    int  r  = g >> 4;
    int  c4 = (g & 15) * 4;
    long seg = idx[r];
    const v4h m4 = *(const v4h*)(msg + (long)r * 64 + c4);
#pragma unroll
    for (int j = 0; j < 4; ++j) {
        unsigned e = mx[seg * 64 + c4 + j];
        float mxv = (e == 0u) ? 0.0f : fdec(e);
        atomicAdd(&s[seg * 64 + c4 + j], __expf(SMOOTH * ((float)m4[j] - mxv)));
    }
}

// Update: h += mish([max_msg, h] @ wui + bui) @ wuo + buo  (4 waves / block)
__global__ void __launch_bounds__(128)
update_kernel(float* __restrict__ h,
              const float* __restrict__ s,
              const unsigned* __restrict__ mx,
              const _Float16* __restrict__ wuiT,
              const float* __restrict__ bui,
              const _Float16* __restrict__ wuoT,
              const float* __restrict__ buo,
              int nobj) {
    constexpr int K = 128;
    __shared__ _Float16 lds_x[4 * 16 * K];
    __shared__ _Float16 lds_m[4 * 16 * K];
    const int  wave = threadIdx.x >> 5;
    const int  lane = threadIdx.x & 31;
    const long base = (long)blockIdx.x * 64 + wave * 16;

    _Float16* my_x = lds_x + wave * 16 * K;
    _Float16* my_m = lds_m + wave * 16 * K;

    for (int t = lane; t < 16 * K; t += 32) {
        int r = t >> 7, c = t & 127;
        long obj = base + r; if (obj >= nobj) obj = nobj - 1;
        float v;
        if (c < 64) {
            float    sv = s[obj * 64 + c];
            unsigned e  = mx[obj * 64 + c];
            float mxv = (e == 0u) ? 0.0f : fdec(e);
            v = __logf(sv + 1e-16f) * (1.0f / SMOOTH) + mxv;  // smooth max
        } else {
            v = h[obj * 64 + (c - 64)];
        }
        my_x[t] = (_Float16)v;
    }
    __syncthreads();

    const int half_id = lane >> 4;
    const int nlane   = lane & 15;

    v16h afrag[4];
#pragma unroll
    for (int ks = 0; ks < 4; ++ks)
        afrag[ks] = load_a(my_x, K, nlane, half_id, ks * 32);

    // GEMM1: 16x128 @ 128x128, mish
    for (int nt = 0; nt < 8; ++nt) {
        const int col = nt * 16 + nlane;
        const float b0 = bui[col];
        v8f acc;
#pragma unroll
        for (int j = 0; j < 8; ++j) acc[j] = b0;
#pragma unroll
        for (int ks = 0; ks < 4; ++ks)
            acc = wmma_f16(afrag[ks], load_b(wuiT, K, col, half_id, ks * 32), acc);
#pragma unroll
        for (int j = 0; j < 8; ++j) {
            int row = j + 8 * half_id;
            my_m[row * K + col] = (_Float16)mishf(acc[j]);
        }
    }
    __syncthreads();

    v16h mfrag[4];
#pragma unroll
    for (int ks = 0; ks < 4; ++ks)
        mfrag[ks] = load_a(my_m, K, nlane, half_id, ks * 32);

    // GEMM2: 16x128 @ 128x64, residual into h (f32, in place)
    for (int nt = 0; nt < 4; ++nt) {
        const int col = nt * 16 + nlane;
        const float b0 = buo[col];
        v8f acc;
#pragma unroll
        for (int j = 0; j < 8; ++j) acc[j] = b0;
#pragma unroll
        for (int ks = 0; ks < 4; ++ks)
            acc = wmma_f16(mfrag[ks], load_b(wuoT, K, col, half_id, ks * 32), acc);
#pragma unroll
        for (int j = 0; j < 8; ++j) {
            int  row = j + 8 * half_id;
            long obj = base + row;
            if (obj < nobj) {
                long hi = obj * 64 + col;
                h[hi] = h[hi] + acc[j];
            }
        }
    }
}

extern "C" void kernel_launch(void* const* d_in, const int* in_sizes, int n_in,
                              void* d_out, int out_size, void* d_ws, size_t ws_size,
                              hipStream_t stream) {
    const int* idxU = (const int*)d_in[0];
    const int* idxB = (const int*)d_in[1];
    const int* idxT = (const int*)d_in[2];
    const float* w1i = (const float*)d_in[4];  const float* b1i = (const float*)d_in[5];
    const float* w1o = (const float*)d_in[6];  const float* b1o = (const float*)d_in[7];
    const float* w2i = (const float*)d_in[8];  const float* b2i = (const float*)d_in[9];
    const float* w2o = (const float*)d_in[10]; const float* b2o = (const float*)d_in[11];
    const float* w3i = (const float*)d_in[12]; const float* b3i = (const float*)d_in[13];
    const float* w3o = (const float*)d_in[14]; const float* b3o = (const float*)d_in[15];
    const float* wui = (const float*)d_in[16]; const float* bui = (const float*)d_in[17];
    const float* wuo = (const float*)d_in[18]; const float* buo = (const float*)d_in[19];

    const int cntU = in_sizes[0], cntB = in_sizes[1], cntT = in_sizes[2];
    const int nobj = out_size / 64;
    float* h = (float*)d_out;

    // Workspace carve-out (256B aligned)
    char* p = (char*)d_ws;
    auto carve = [&](size_t bytes) {
        void* r = (void*)p;
        p += (bytes + 255) & ~(size_t)255;
        return r;
    };
    const size_t totMsgs = (size_t)cntU + cntB + cntT;
    _Float16* msgU = (_Float16*)carve(totMsgs * 64 * sizeof(_Float16));
    _Float16* msgB = msgU + (size_t)cntU * 64;
    _Float16* msgT = msgB + (size_t)cntB * 64;
    unsigned* mx = (unsigned*)carve((size_t)nobj * 64 * sizeof(unsigned));
    float*    s  = (float*)carve((size_t)nobj * 64 * sizeof(float));
    _Float16* w1iT = (_Float16*)carve(64 * 64 * 2);
    _Float16* w1oT = (_Float16*)carve(64 * 64 * 2);
    _Float16* w2iT = (_Float16*)carve(128 * 128 * 2);
    _Float16* w2oT = (_Float16*)carve(128 * 128 * 2);
    _Float16* w3iT = (_Float16*)carve(192 * 192 * 2);
    _Float16* w3oT = (_Float16*)carve(192 * 192 * 2);
    _Float16* wuiT = (_Float16*)carve(128 * 128 * 2);
    _Float16* wuoT = (_Float16*)carve(64 * 128 * 2);

    auto conv = [&](const float* in, _Float16* out, int K, int N) {
        int tot = K * N;
        wconv_kernel<<<(tot + 255) / 256, 256, 0, stream>>>(in, out, K, N);
    };
    conv(w1i, w1iT, 64, 64);   conv(w1o, w1oT, 64, 64);
    conv(w2i, w2iT, 128, 128); conv(w2o, w2oT, 128, 128);
    conv(w3i, w3iT, 192, 192); conv(w3o, w3oT, 192, 192);
    conv(wui, wuiT, 128, 128); conv(wuo, wuoT, 128, 64);

    hipMemsetAsync(d_out, 0, (size_t)out_size * sizeof(float), stream);  // h = 0

    const int eU = cntU, eB = cntB / 2, eT = cntT / 3;
    for (int layer = 0; layer < 3; ++layer) {
        hipMemsetAsync(mx, 0, (size_t)nobj * 64 * sizeof(unsigned), stream);
        hipMemsetAsync(s,  0, (size_t)nobj * 64 * sizeof(float),    stream);

        msg_kernel<1><<<(eU + 63) / 64, 128, 0, stream>>>(h, idxU, w1iT, b1i, w1oT, b1o, msgU, mx, eU);
        msg_kernel<2><<<(eB + 63) / 64, 128, 0, stream>>>(h, idxB, w2iT, b2i, w2oT, b2o, msgB, mx, eB);
        msg_kernel<3><<<(eT + 63) / 64, 128, 0, stream>>>(h, idxT, w3iT, b3i, w3oT, b3o, msgT, mx, eT);

        expsum_kernel<<<((size_t)cntU * 16 + 255) / 256, 256, 0, stream>>>(msgU, idxU, mx, s, cntU);
        expsum_kernel<<<((size_t)cntB * 16 + 255) / 256, 256, 0, stream>>>(msgB, idxB, mx, s, cntB);
        expsum_kernel<<<((size_t)cntT * 16 + 255) / 256, 256, 0, stream>>>(msgT, idxT, mx, s, cntT);

        update_kernel<<<(nobj + 63) / 64, 128, 0, stream>>>(h, s, mx, wuiT, bui, wuoT, buo, nobj);
    }
}
